// NN_mlpc_qc_1460288881271
// MI455X (gfx1250) — compile-verified, matrix operand order checked by reading, not run on hardware
//
#include <hip/hip_runtime.h>
#include <hip/hip_bf16.h>
#include <math.h>

// ---------------------------------------------------------------------------
// B=4096 OBS=128 L=4 D=8 BINS=16 HID=512 S=32. Compute-bound fused MLPs on
// v_wmma_f32_16x16x32_f16 with 2x2 register tile blocking.
// ---------------------------------------------------------------------------
typedef _Float16 f16;
typedef _Float16 v16h __attribute__((ext_vector_type(16)));
typedef _Float16 v8h  __attribute__((ext_vector_type(8)));
typedef float    v8f  __attribute__((ext_vector_type(8)));

#define B_N    4096
#define OBS_N  128
#define L_N    4
#define D_N    8
#define BINS_N 16
#define HID_N  512
#define S_N    32

#define ROWS    64          // batch rows per workgroup
#define X0STRQ  168         // padded LDS row stride (f16) for Q input (136 real)
#define X0STRV  136         // padded LDS row stride (f16) for V input (128 real)
#define HSTR    520         // padded LDS row stride (f16) for hidden (512 real)

// Branch-free tanh: use V_TANH_F32 if the builtin exists, else
// tanh(x) = 1 - 2/(exp2(2*log2e*x)+1)  (saturates correctly, no exec churn).
__device__ __forceinline__ float fast_tanh(float x) {
#if __has_builtin(__builtin_amdgcn_tanhf)
    return __builtin_amdgcn_tanhf(x);
#else
    float t = __builtin_amdgcn_exp2f(x * 2.8853900817779268f);
    return 1.0f - 2.0f * __builtin_amdgcn_rcpf(t + 1.0f);
#endif
}

// ---------------------------------------------------------------------------
// Weight pre-pass: f32 (K x N row-major) -> f16 WMMA-B fragment-major layout.
// Fragment (colTile c, kTile kt): 32 lanes x 16 halves contiguous; K mapping
// per ISA 7.12.2: khalf = lane>>4 ; elem i: K = kt*32 + (i<8?i:i+8) + 8*khalf
// ---------------------------------------------------------------------------
__global__ void conv_wfrag(const float* __restrict__ src, f16* __restrict__ dst,
                           int Kreal, int Kpad, int N) {
    int KT = Kpad >> 5;
    int total = N * Kpad;
    for (int idx = blockIdx.x * blockDim.x + threadIdx.x; idx < total;
         idx += gridDim.x * blockDim.x) {
        int i    = idx & 15;
        int lane = (idx >> 4) & 31;
        int ckt  = idx >> 9;
        int kt   = ckt % KT;
        int c    = ckt / KT;
        int n    = (c << 4) + (lane & 15);
        int kh   = lane >> 4;
        int k    = (kt << 5) + (i < 8 ? i : i + 8) + (kh << 3);
        float v  = (k < Kreal) ? src[k * N + n] : 0.0f;
        dst[idx] = (f16)v;
    }
}

__global__ void zero_fill(float* __restrict__ p, int n) {
    int i = blockIdx.x * blockDim.x + threadIdx.x;
    if (i < n) p[i] = 0.0f;
}

// LDS A-fragment (16-bit A 16x32, ISA 7.12.2): two contiguous 16B ds loads.
__device__ __forceinline__ v16h ldsA(const f16* __restrict__ base, int stride,
                                     int row, int k0, int kh) {
    const f16* p = base + row * stride + k0 + (kh << 3);
    v8h lo = *(const v8h*)(p);
    v8h hi = *(const v8h*)(p + 16);
    return __builtin_shufflevector(lo, hi, 0, 1, 2, 3, 4, 5, 6, 7,
                                   8, 9, 10, 11, 12, 13, 14, 15);
}

#define WMMA_F16(a, b, c) \
    __builtin_amdgcn_wmma_f32_16x16x32_f16(false, (a), false, (b), (short)0, (c), false, false)

// Store one 16x16 f32 accumulator tile to LDS f16 (C/D layout: lane holds col
// lane&15, rows v + 8*(lane>>4)).
template <bool TANH>
__device__ __forceinline__ void store_tile(v8f acc, int rt, int ct,
                                           f16* __restrict__ out,
                                           const float* __restrict__ bias,
                                           int m, int kh) {
    float bb = bias[(ct << 4) + m];
    f16* o = out + (ct << 4) + m;
#pragma unroll
    for (int v = 0; v < 8; ++v) {
        float h = acc[v] + bb;
        if (TANH) h = fast_tanh(h);
        o[((rt << 4) + v + (kh << 3)) * HSTR] = (f16)h;
    }
}

// ---------------------------------------------------------------------------
// One fused layer with 2x2 tile blocking: in (ROWS x K, LDS f16) @ W
// (K x 512, frag f16) -> out (ROWS x 512, LDS f16). 8 waves x 4 blocks each;
// each block = 2 row-tiles x 2 col-tiles -> 4 WMMAs per (2 A + 2 B) loads.
// ---------------------------------------------------------------------------
template <int KT, bool TANH>
__device__ __forceinline__ void gemm_layer(const f16* __restrict__ in, int instr,
                                           f16* __restrict__ out,
                                           const f16* __restrict__ wfrag,
                                           const float* __restrict__ bias,
                                           int wave, int lane) {
    const int m = lane & 15, kh = lane >> 4;
    for (int blk = wave; blk < 32; blk += 8) {
        int rt0 = (blk & 1) << 1, rt1 = rt0 + 1;     // row tiles 0..3
        int ct0 = (blk >> 1) << 1, ct1 = ct0 + 1;    // col tiles 0..31
        v8f acc00 = {0.f, 0.f, 0.f, 0.f, 0.f, 0.f, 0.f, 0.f};
        v8f acc01 = acc00, acc10 = acc00, acc11 = acc00;
        const f16* wp0 = wfrag + (size_t)(ct0 * KT) * 512 + lane * 16;
        const f16* wp1 = wp0 + (size_t)KT * 512;
#pragma unroll 2
        for (int kt = 0; kt < KT; ++kt) {
            v16h a0 = ldsA(in, instr, (rt0 << 4) + m, kt << 5, kh);
            v16h a1 = ldsA(in, instr, (rt1 << 4) + m, kt << 5, kh);
            v16h b0 = *(const v16h*)wp0; wp0 += 512;
            v16h b1 = *(const v16h*)wp1; wp1 += 512;
            acc00 = WMMA_F16(a0, b0, acc00);
            acc01 = WMMA_F16(a0, b1, acc01);
            acc10 = WMMA_F16(a1, b0, acc10);
            acc11 = WMMA_F16(a1, b1, acc11);
        }
        store_tile<TANH>(acc00, rt0, ct0, out, bias, m, kh);
        store_tile<TANH>(acc01, rt0, ct1, out, bias, m, kh);
        store_tile<TANH>(acc10, rt1, ct0, out, bias, m, kh);
        store_tile<TANH>(acc11, rt1, ct1, out, bias, m, kh);
    }
}

// actions_from_category, evaluated on the fly per (b, s, d).
__device__ __forceinline__ float action_val(const int* __restrict__ cat,
                                            int b, int s, int d) {
    int lvl = (s >> 3) + ((d < (s & 7)) ? 1 : 0);
    float wj = 2.0f / 16.0f;
    float cum = 0.0f;
    for (int j = 0; j < lvl; ++j) {
        cum += (float)cat[(b * L_N + j) * D_N + d] * wj;
        if (j + 1 == lvl) return -1.0f + cum + 0.5f * wj;
        wj *= 0.0625f;
    }
    return 0.0f;
}

// ---------------------------------------------------------------------------
// Q network: 131072 rows of [obs(128)|actions(8)] -> 3x tanh(512) -> diag(16)
// LDS: xb(64x168) + hA(64x520) + hB(64x520) f16 = 154624 bytes
// ---------------------------------------------------------------------------
__global__ __launch_bounds__(256) void q_mlp_kernel(
    const float* __restrict__ obs, const int* __restrict__ cat,
    const f16* __restrict__ w0f, const f16* __restrict__ w1f,
    const f16* __restrict__ w2f, const f16* __restrict__ w3f,
    const float* __restrict__ b0, const float* __restrict__ b1,
    const float* __restrict__ b2, const float* __restrict__ b3,
    float* __restrict__ adv) {
    extern __shared__ f16 smem[];
    f16* xb = smem;                      // 64*168
    f16* hA = smem + ROWS * X0STRQ;      // 64*520
    f16* hB = hA + ROWS * HSTR;          // 64*520

    const int tid = threadIdx.x;
    const int wave = tid >> 5, lane = tid & 31;

    // Stage X0 = [obs | actions | zero-pad]
    for (int idx = tid; idx < ROWS * X0STRQ; idx += 256) {
        int row = idx / X0STRQ, c = idx % X0STRQ;
        int r = blockIdx.x * ROWS + row;
        int b = r >> 5, s = r & 31;
        float val = 0.0f;
        if (c < 128)      val = obs[b * OBS_N + c];
        else if (c < 136) val = action_val(cat, b, s, c - 128);
        xb[row * X0STRQ + c] = (f16)val;
    }
    __syncthreads();

    gemm_layer<5, true>(xb, X0STRQ, hA, w0f, b0, wave, lane);   // 136(->160)->512
    __syncthreads();
    gemm_layer<16, true>(hA, HSTR, hB, w1f, b1, wave, lane);    // 512->512
    __syncthreads();
    gemm_layer<16, true>(hB, HSTR, hA, w2f, b2, wave, lane);    // 512->512
    __syncthreads();

    // Final layer: row-tile rt only needs col-tiles ct = (rt&1)*16 + ci, and
    // within tile (rt,ct) only row m == ci matters (diagonal extraction).
    {
        const int m = lane & 15, kh = lane >> 4;
        for (int t = wave; t < 64; t += 8) {
            int rt = t >> 4;
            int ci = t & 15;
            int ct = ((rt & 1) << 4) + ci;       // equals s of the needed row
            v8f acc = {0.f, 0.f, 0.f, 0.f, 0.f, 0.f, 0.f, 0.f};
            const f16* wp = w3f + (size_t)(ct * 16) * 512 + lane * 16;
#pragma unroll 4
            for (int kt = 0; kt < 16; ++kt) {
                v16h a = ldsA(hA, HSTR, (rt << 4) + m, kt << 5, kh);
                v16h b = *(const v16h*)wp;
                wp += 512;
                acc = WMMA_F16(a, b, acc);
            }
            if (kh == (ci >> 3)) {   // 16 lanes hold row m=ci, cols = lane&15
                int r = blockIdx.x * ROWS + (rt << 4) + ci;
                int bidx = r >> 5;
                float val = acc[ci & 7] + b3[(ct << 4) + m];
                adv[(size_t)bidx * 512 + (ct << 4) + m] = val;
            }
        }
    }
}

// ---------------------------------------------------------------------------
// V network: 4096 rows of obs -> 3x tanh(512) -> 1
// LDS: xb(64x136) + hA(64x520) + hB(64x520) f16 = 150528 bytes
// ---------------------------------------------------------------------------
__global__ __launch_bounds__(256) void v_mlp_kernel(
    const float* __restrict__ obs,
    const f16* __restrict__ w0f, const f16* __restrict__ w1f,
    const f16* __restrict__ w2f,
    const float* __restrict__ b0, const float* __restrict__ b1,
    const float* __restrict__ b2,
    const float* __restrict__ w3, const float* __restrict__ b3,
    float* __restrict__ value) {
    extern __shared__ f16 smem[];
    f16* xb = smem;                      // 64*136
    f16* hA = smem + ROWS * X0STRV;      // 64*520
    f16* hB = hA + ROWS * HSTR;          // 64*520

    const int tid = threadIdx.x;
    const int wave = tid >> 5, lane = tid & 31;

    for (int idx = tid; idx < ROWS * X0STRV; idx += 256) {
        int row = idx / X0STRV, c = idx % X0STRV;
        int r = blockIdx.x * ROWS + row;
        float val = (c < 128) ? obs[r * OBS_N + c] : 0.0f;
        xb[row * X0STRV + c] = (f16)val;
    }
    __syncthreads();

    gemm_layer<4, true>(xb, X0STRV, hA, w0f, b0, wave, lane);   // 128->512
    __syncthreads();
    gemm_layer<16, true>(hA, HSTR, hB, w1f, b1, wave, lane);    // 512->512
    __syncthreads();
    gemm_layer<16, true>(hB, HSTR, hA, w2f, b2, wave, lane);    // 512->512
    __syncthreads();

    // 512 -> 1 head: 4 threads per row, shuffle-reduce.
    {
        int row = tid >> 2, part = tid & 3;
        float sum = 0.0f;
        const f16* hp = hA + row * HSTR + part * 128;
        const float* wp = w3 + part * 128;
#pragma unroll 8
        for (int c = 0; c < 128; ++c) sum += (float)hp[c] * wp[c];
        sum += __shfl_xor(sum, 1, 32);
        sum += __shfl_xor(sum, 2, 32);
        if (part == 0) {
            int r = blockIdx.x * ROWS + row;
            value[r] = sum + b3[0];
        }
    }
}

// ---------------------------------------------------------------------------
// Host launch
// ---------------------------------------------------------------------------
extern "C" void kernel_launch(void* const* d_in, const int* in_sizes, int n_in,
                              void* d_out, int out_size, void* d_ws, size_t ws_size,
                              hipStream_t stream) {
    (void)in_sizes; (void)n_in; (void)out_size; (void)ws_size;
    const float* obs = (const float*)d_in[0];
    const int*   cat = (const int*)d_in[1];
    const float* vw0 = (const float*)d_in[2];  const float* vb0 = (const float*)d_in[3];
    const float* vw1 = (const float*)d_in[4];  const float* vb1 = (const float*)d_in[5];
    const float* vw2 = (const float*)d_in[6];  const float* vb2 = (const float*)d_in[7];
    const float* vw3 = (const float*)d_in[8];  const float* vb3 = (const float*)d_in[9];
    const float* qw0 = (const float*)d_in[10]; const float* qb0 = (const float*)d_in[11];
    const float* qw1 = (const float*)d_in[12]; const float* qb1 = (const float*)d_in[13];
    const float* qw2 = (const float*)d_in[14]; const float* qb2 = (const float*)d_in[15];
    const float* qw3 = (const float*)d_in[16]; const float* qb3 = (const float*)d_in[17];

    // Workspace layout (f16 fragments), ~2.9 MB total.
    f16* ws  = (f16*)d_ws;
    f16* q0f = ws;                      // 160*512
    f16* q1f = q0f + 160 * 512;        // 512*512
    f16* q2f = q1f + 512 * 512;
    f16* q3f = q2f + 512 * 512;
    f16* v0f = q3f + 512 * 512;        // 128*512
    f16* v1f = v0f + 128 * 512;
    f16* v2f = v1f + 512 * 512;

    conv_wfrag<<<320, 256, 0, stream>>>(qw0, q0f, 136, 160, 512);
    conv_wfrag<<<1024, 256, 0, stream>>>(qw1, q1f, 512, 512, 512);
    conv_wfrag<<<1024, 256, 0, stream>>>(qw2, q2f, 512, 512, 512);
    conv_wfrag<<<1024, 256, 0, stream>>>(qw3, q3f, 512, 512, 512);
    conv_wfrag<<<256, 256, 0, stream>>>(vw0, v0f, 128, 128, 512);
    conv_wfrag<<<1024, 256, 0, stream>>>(vw1, v1f, 512, 512, 512);
    conv_wfrag<<<1024, 256, 0, stream>>>(vw2, v2f, 512, 512, 512);

    float* out_value = (float*)d_out;                 // 4096
    float* out_adv   = out_value + 4096;              // 4096*512
    float* out_nbias = out_adv + 4096 * 512;          // 4096*32

    const int q_shmem = (ROWS * X0STRQ + 2 * ROWS * HSTR) * (int)sizeof(f16); // 154624
    const int v_shmem = (ROWS * X0STRV + 2 * ROWS * HSTR) * (int)sizeof(f16); // 150528
    hipFuncSetAttribute(reinterpret_cast<const void*>(q_mlp_kernel),
                        hipFuncAttributeMaxDynamicSharedMemorySize, q_shmem);
    hipFuncSetAttribute(reinterpret_cast<const void*>(v_mlp_kernel),
                        hipFuncAttributeMaxDynamicSharedMemorySize, v_shmem);

    q_mlp_kernel<<<(B_N * S_N) / ROWS, 256, q_shmem, stream>>>(
        obs, cat, q0f, q1f, q2f, q3f, qb0, qb1, qb2, qb3, out_adv);
    v_mlp_kernel<<<B_N / ROWS, 256, v_shmem, stream>>>(
        obs, v0f, v1f, v2f, vb0, vb1, vb2, vw3, vb3, out_value);
    zero_fill<<<(4096 * 32 + 255) / 256, 256, 0, stream>>>(out_nbias, 4096 * 32);
}
